// MFIB_8985071583602
// MI455X (gfx1250) — matrix operand order repaired and checked by reading, not compiled, not run on hardware
//
#include <hip/hip_runtime.h>
#include <hip/hip_bf16.h>
#include <math.h>

// ---------------------------------------------------------------------------
// MI455X (gfx1250): all contractions via v_wmma_f32_16x16x32_f16, LDS operands
// laid out so every fragment load is two 16B (ds_load_b128) loads.
// ---------------------------------------------------------------------------

typedef __attribute__((ext_vector_type(16))) _Float16 v16h;
typedef __attribute__((ext_vector_type(8)))  _Float16 v8h;
typedef __attribute__((ext_vector_type(8)))  float    v8f;

#define DEVINL __device__ __forceinline__

constexpr int NB  = 8;      // batch
constexpr int NC  = 96;     // channels
constexpr int NOC = 192;    // 2C qkv channels
constexpr int NH  = 128;
constexpr int NW  = 128;
constexpr int NN  = NH * NW;   // 16384

// ----- workspace layout (bytes) -----
constexpr size_t OFF_STATS = 0;                              // 64 f32
constexpr size_t OFF_WH    = 256;                            // f16 [2][192][96]
constexpr size_t OFF_BIASQ = OFF_WH + (size_t)2*NOC*NC*2;    // f32 [2][8][192]
constexpr size_t OFF_RESC  = OFF_BIASQ + (size_t)2*NB*NOC*4; // f32 [2][2][8][96]
constexpr size_t OFF_AC    = OFF_RESC + (size_t)4*NB*NC*4;   // f16 [8][32][32]
constexpr size_t OFF_QKVU  = (OFF_AC + (size_t)NB*32*32*2 + 255) & ~(size_t)255;
constexpr size_t SZ_QKV    = (size_t)NB*NOC*NN*2;            // f16
constexpr size_t OFF_QKVD  = OFF_QKVU + SZ_QKV;
constexpr size_t SZ_ATT    = (size_t)NB*NC*NN*4;             // f32
constexpr size_t OFF_D2U   = OFF_QKVD + SZ_QKV;
constexpr size_t OFF_U2D   = OFF_D2U + SZ_ATT;

// ----- WMMA helpers --------------------------------------------------------
DEVINL v8f wmma_f16(v16h a, v16h b, v8f c) {
  return __builtin_amdgcn_wmma_f32_16x16x32_f16(
      /*neg_a=*/false, a, /*neg_b=*/false, b,
      /*c_mod=*/(short)0, c, /*reuse_a=*/false, /*reuse_b=*/false);
}

union F16x16 {
  struct { uint4 lo, hi; } q;
  v16h v;
};

// A-fragment (split-K pattern): elems 0..7 at p[0..7], elems 8..15 at p[16..23]
// caller passes p = base + row*ld + kb + lhA  (16B aligned)
DEVINL v16h fragA(const _Float16* p) {
  F16x16 u;
  u.q.lo = *(const uint4*)p;
  u.q.hi = *(const uint4*)(p + 16);
  return u.v;
}
// B-fragment (split-K pattern): elems 0..7 at p[0..7], elems 8..15 at p[16..23]
// caller passes p = base + row*ld + kb + lhB  (16B aligned)
DEVINL v16h fragB(const _Float16* p) {
  F16x16 u;
  u.q.lo = *(const uint4*)p;
  u.q.hi = *(const uint4*)(p + 16);
  return u.v;
}
// Softmax-on-the-fly A-fragment: row stats fixed per lane.
DEVINL v16h fragP(const _Float16* p, float m, float inv) {
  v8h s0 = *(const v8h*)p;
  v8h s1 = *(const v8h*)(p + 16);
  v16h a;
#pragma unroll
  for (int t = 0; t < 8; ++t) {
    a[t]     = (_Float16)(__expf((float)s0[t] - m) * inv);
    a[8 + t] = (_Float16)(__expf((float)s1[t] - m) * inv);
  }
  return a;
}
// Softmax-on-the-fly A-fragment with per-element stats (transposed P).
DEVINL v16h fragPT(const _Float16* p, const float* mx, const float* iv) {
  v8h s0 = *(const v8h*)p;
  v8h s1 = *(const v8h*)(p + 16);
  v16h a;
#pragma unroll
  for (int t = 0; t < 8; ++t) {
    a[t]     = (_Float16)(__expf((float)s0[t] - mx[t]) * iv[t]);
    a[8 + t] = (_Float16)(__expf((float)s1[t] - mx[16 + t]) * iv[16 + t]);
  }
  return a;
}

// ===========================================================================
// Kernel 1: per-(sample,side) mean / std / 1/std over C*H*W
// ===========================================================================
__global__ __launch_bounds__(256) void k_stats(const float* __restrict__ x_u,
                                               const float* __restrict__ x_d,
                                               float* __restrict__ stats) {
  int sb = blockIdx.x, side = sb >> 3, b = sb & 7;
  const float* xg = (side ? x_d : x_u) + (size_t)b * NC * NN;
  const size_t M4 = (size_t)NC * NN / 4;
  const float4* xg4 = (const float4*)xg;
  float s1 = 0.f, s2 = 0.f;
  for (size_t i = threadIdx.x; i < M4; i += 256) {
    float4 v = xg4[i];
    s1 += v.x + v.y + v.z + v.w;
    s2 += v.x * v.x + v.y * v.y + v.z * v.z + v.w * v.w;
  }
  __shared__ float r1[256], r2[256];
  r1[threadIdx.x] = s1; r2[threadIdx.x] = s2;
  __syncthreads();
  for (int o = 128; o > 0; o >>= 1) {
    if ((int)threadIdx.x < o) {
      r1[threadIdx.x] += r1[threadIdx.x + o];
      r2[threadIdx.x] += r2[threadIdx.x + o];
    }
    __syncthreads();
  }
  if (threadIdx.x == 0) {
    const float M = (float)((size_t)NC * NN);
    float mean = r1[0] / M;
    float var  = r2[0] / M - mean * mean;
    float stdv = sqrtf(var + 1e-5f);
    float* sp = stats + side * 32;
    sp[b] = mean; sp[8 + b] = stdv; sp[16 + b] = 1.f / stdv;
  }
}

// ===========================================================================
// Kernel 2: fold RLN affine into qkv weights; per-(b,o) bias; rescale/rebias
// ===========================================================================
__global__ void k_prep(const float* __restrict__ Wq_u, const float* __restrict__ bq_u,
                       const float* __restrict__ Wq_d, const float* __restrict__ bq_d,
                       const float* __restrict__ nw_u, const float* __restrict__ nb_u,
                       const float* __restrict__ m1w_u, const float* __restrict__ m1b_u,
                       const float* __restrict__ m2w_u, const float* __restrict__ m2b_u,
                       const float* __restrict__ nw_d, const float* __restrict__ nb_d,
                       const float* __restrict__ m1w_d, const float* __restrict__ m1b_d,
                       const float* __restrict__ m2w_d, const float* __restrict__ m2b_d,
                       const float* __restrict__ stats,
                       _Float16* __restrict__ Wh, float* __restrict__ biasQ,
                       float* __restrict__ resc) {
  int side = blockIdx.x;
  const float* W   = side ? Wq_d : Wq_u;
  const float* bq  = side ? bq_d : bq_u;
  const float* nw  = side ? nw_d : nw_u;
  const float* nb  = side ? nb_d : nb_u;
  const float* m1w = side ? m1w_d : m1w_u;
  const float* m1b = side ? m1b_d : m1b_u;
  const float* m2w = side ? m2w_d : m2w_u;
  const float* m2b = side ? m2b_d : m2b_u;
  int o = threadIdx.x;  // 0..191
  float s1 = 0.f, s2 = 0.f;
  for (int i = 0; i < NC; ++i) {
    float w  = W[o * NC + i];
    float wp = w * nw[i];
    Wh[((size_t)side * NOC + o) * NC + i] = (_Float16)wp;
    s1 += wp;
    s2 += w * nb[i];
  }
  const float* sp = stats + side * 32;
  for (int b = 0; b < NB; ++b)
    biasQ[((size_t)side * NB + b) * NOC + o] = bq[o] + s2 - sp[b] * sp[16 + b] * s1;
  if (o < NC) {
    for (int b = 0; b < NB; ++b) {
      float mean = sp[b], stdv = sp[8 + b];
      resc[((size_t)(side * 2 + 0) * NB + b) * NC + o] = stdv * m1w[o] + m1b[o];
      resc[((size_t)(side * 2 + 1) * NB + b) * NC + o] = mean * m2w[o] + m2b[o];
    }
  }
}

// ===========================================================================
// Kernel 3: qkv = rstd * (W' @ x) + bias  -> f16     [192 x 96] x [96 x 16384]
// Block tile: 64(M) x 128(N), 8 waves = 4x2, K = 96 = 3 WMMA steps.
// X tile stored TRANSPOSED in LDS so B-fragments are contiguous b128 loads.
// ===========================================================================
__global__ __launch_bounds__(256) void k_qkv_gemm(
    const float* __restrict__ x_u, const float* __restrict__ x_d,
    const _Float16* __restrict__ Wh, const float* __restrict__ biasQ,
    const float* __restrict__ stats,
    _Float16* __restrict__ qkv_u, _Float16* __restrict__ qkv_d) {
  int nBase = blockIdx.x * 128;
  int mBase = blockIdx.y * 64;
  int sb = blockIdx.z, side = sb >> 3, b = sb & 7;
  const float* xg = (side ? x_d : x_u) + (size_t)b * NC * NN;
  _Float16* yg = (side ? qkv_d : qkv_u) + (size_t)b * NOC * NN;
  const _Float16* Wg = Wh + (size_t)side * NOC * NC;
  const float* bg = biasQ + ((size_t)side * NB + b) * NOC;
  float rstd = stats[side * 32 + 16 + b];

  __shared__ __align__(16) _Float16 Wl[64][104];    // [m][k]
  __shared__ __align__(16) _Float16 XlT[128][104];  // [n][k]

  int tid = threadIdx.x;
  for (int idx = tid; idx < 64 * 96; idx += 256) {
    int r = idx / 96, k = idx - r * 96;
    Wl[r][k] = Wg[(size_t)(mBase + r) * NC + k];
  }
  for (int idx = tid; idx < 96 * 128; idx += 256) {
    int r = idx >> 7, n = idx & 127;
    XlT[n][r] = (_Float16)xg[(size_t)r * NN + nBase + n];
  }
  __syncthreads();

  int lane = tid & 31, wv = tid >> 5;
  int lr = lane & 15, lhA = (lane >> 4) << 3, lhB = (lane >> 4) << 3, mo = lhA;
  int wm = wv >> 1, wn = wv & 1;
  int mT = wm * 16;

  v8f acc[4] = {};
#pragma unroll
  for (int kk = 0; kk < 96; kk += 32) {
    v16h a = fragA(&Wl[mT + lr][kk + lhA]);
#pragma unroll
    for (int nt = 0; nt < 4; ++nt) {
      v16h bf = fragB(&XlT[wn * 64 + nt * 16 + lr][kk + lhB]);
      acc[nt] = wmma_f16(a, bf, acc[nt]);
    }
  }
#pragma unroll
  for (int nt = 0; nt < 4; ++nt) {
    int n = nBase + wn * 64 + nt * 16 + lr;
#pragma unroll
    for (int v = 0; v < 8; ++v) {
      int m = mBase + mT + v + mo;
      yg[(size_t)m * NN + n] = (_Float16)(rstd * acc[nt][v] + bg[m]);
    }
  }
}

// ===========================================================================
// Kernel 4: height-axis attention, one block per (b,w).
// Scores kept in BOTH orientations (Sl row-major, SlT col-major) so both
// P and P^T fragments are contiguous; softmax is regenerated with __expf.
// ===========================================================================
__global__ __launch_bounds__(256) void k_attn_h(
    const _Float16* __restrict__ qkv_u, const _Float16* __restrict__ qkv_d,
    const float* __restrict__ scale_h,
    float* __restrict__ d2u, float* __restrict__ u2d) {
  int b = blockIdx.x >> 7;
  int w = blockIdx.x & 127;
  const _Float16* Qg  = qkv_u + (size_t)b * NOC * NN;                // ch 0..31
  const _Float16* Kg  = qkv_d + (size_t)b * NOC * NN;                // ch 0..31
  const _Float16* Vdg = Kg + (size_t)96 * NN;                        // ch 96..127
  const _Float16* Vug = Qg + (size_t)96 * NN;

  __shared__ __align__(16) _Float16 KlT[128][32];   // [j][c]
  __shared__ __align__(16) _Float16 Vdl[32][128];   // [c][j]
  __shared__ __align__(16) _Float16 Vul[32][128];   // [c][i]
  __shared__ __align__(16) _Float16 Sl [128][128];  // [i][j]
  __shared__ __align__(16) _Float16 SlT[128][128];  // [j][i]
  __shared__ __align__(16) float rmax[128], rinv[128];

  int tid = threadIdx.x;
  for (int idx = tid; idx < 32 * 128; idx += 256) {
    int c = idx >> 7, i = idx & 127;
    size_t g = (size_t)c * NN + (size_t)i * NW + w;
    KlT[i][c] = Kg[g]; Vdl[c][i] = Vdg[g]; Vul[c][i] = Vug[g];
  }
  __syncthreads();

  int lane = tid & 31, wv = tid >> 5;
  int lr = lane & 15, lhA = (lane >> 4) << 3, lhB = (lane >> 4) << 3, mo = lhA;
  float sc = scale_h[0];

  {  // scores: wave wv handles i-tile wv
    int i = wv * 16 + lr;
    v16h a;
#pragma unroll
    for (int t = 0; t < 8; ++t) a[t]     = Qg[(size_t)(lhA + t) * NN + (size_t)i * NW + w];
#pragma unroll
    for (int t = 0; t < 8; ++t) a[8 + t] = Qg[(size_t)(16 + lhA + t) * NN + (size_t)i * NW + w];
#pragma unroll
    for (int jt = 0; jt < 8; ++jt) {
      int j = jt * 16 + lr;
      v16h bf = fragB(&KlT[j][lhB]);
      v8f acc = {};
      acc = wmma_f16(a, bf, acc);
      v8h sv;
#pragma unroll
      for (int v = 0; v < 8; ++v) sv[v] = (_Float16)(acc[v] * sc);
      *(v8h*)&SlT[j][wv * 16 + mo] = sv;            // contiguous 16B store
#pragma unroll
      for (int v = 0; v < 8; ++v) Sl[wv * 16 + v + mo][j] = sv[v];
    }
  }
  __syncthreads();

  if (tid < 128) {  // row softmax stats (contiguous v8h reads of Sl row)
    float m = -1e30f;
    for (int j = 0; j < 128; j += 8) {
      v8h s = *(const v8h*)&Sl[tid][j];
#pragma unroll
      for (int t = 0; t < 8; ++t) m = fmaxf(m, (float)s[t]);
    }
    float s = 0.f;
    for (int j = 0; j < 128; j += 8) {
      v8h sv = *(const v8h*)&Sl[tid][j];
#pragma unroll
      for (int t = 0; t < 8; ++t) s += __expf((float)sv[t] - m);
    }
    rmax[tid] = m; rinv[tid] = 1.f / s;
  }
  __syncthreads();

  {  // O1[i,c] = sum_j P[i,j] Vd[c,j]
    v8f acc[2] = {};
    int i = wv * 16 + lr;
    float m = rmax[i], inv = rinv[i];
#pragma unroll
    for (int kk = 0; kk < 128; kk += 32) {
      v16h a = fragP(&Sl[i][kk + lhA], m, inv);
#pragma unroll
      for (int ct = 0; ct < 2; ++ct) {
        v16h bf = fragB(&Vdl[ct * 16 + lr][kk + lhB]);
        acc[ct] = wmma_f16(a, bf, acc[ct]);
      }
    }
    float* og = d2u + (size_t)b * NC * NN;
#pragma unroll
    for (int ct = 0; ct < 2; ++ct) {
      int c = ct * 16 + lr;
#pragma unroll
      for (int v = 0; v < 8; ++v) {
        int ii = wv * 16 + v + mo;
        og[(size_t)c * NN + (size_t)ii * NW + w] = acc[ct][v];
      }
    }
  }

  {  // O2[j,c] = sum_i P[i,j] Vu[c,i]   (P^T fragments from SlT, contiguous)
    v8f acc[2] = {};
    int j = wv * 16 + lr;
#pragma unroll
    for (int kk = 0; kk < 128; kk += 32) {
      v16h a = fragPT(&SlT[j][kk + lhA], &rmax[kk + lhA], &rinv[kk + lhA]);
#pragma unroll
      for (int ct = 0; ct < 2; ++ct) {
        v16h bf = fragB(&Vul[ct * 16 + lr][kk + lhB]);
        acc[ct] = wmma_f16(a, bf, acc[ct]);
      }
    }
    float* og = u2d + (size_t)b * NC * NN;
#pragma unroll
    for (int ct = 0; ct < 2; ++ct) {
      int c = ct * 16 + lr;
#pragma unroll
      for (int v = 0; v < 8; ++v) {
        int jj = wv * 16 + v + mo;
        og[(size_t)c * NN + (size_t)jj * NW + w] = acc[ct][v];
      }
    }
  }
}

// ===========================================================================
// Kernel 5: width-axis attention, one block per (b,h). Row AND column softmax.
// ===========================================================================
__global__ __launch_bounds__(256) void k_attn_w(
    const _Float16* __restrict__ qkv_u, const _Float16* __restrict__ qkv_d,
    const float* __restrict__ scale_w,
    float* __restrict__ d2u, float* __restrict__ u2d) {
  int b = blockIdx.x >> 7;
  int h = blockIdx.x & 127;
  const _Float16* Qg  = qkv_u + ((size_t)b * NOC + 32) * NN;   // q_w
  const _Float16* Kg  = qkv_d + ((size_t)b * NOC + 32) * NN;   // k_w
  const _Float16* Vdg = qkv_d + ((size_t)b * NOC + 128) * NN;  // vd_w
  const _Float16* Vug = qkv_u + ((size_t)b * NOC + 128) * NN;  // vu_w

  __shared__ __align__(16) _Float16 KlT[128][32];
  __shared__ __align__(16) _Float16 Vdl[32][128];
  __shared__ __align__(16) _Float16 Vul[32][128];
  __shared__ __align__(16) _Float16 Sl [128][128];
  __shared__ __align__(16) _Float16 SlT[128][128];
  __shared__ __align__(16) float rmax[128], rinv[128], cmax[128], cinv[128];

  int tid = threadIdx.x;
  size_t rowoff = (size_t)h * NW;
  for (int idx = tid; idx < 32 * 128; idx += 256) {
    int c = idx >> 7, i = idx & 127;
    size_t g = (size_t)c * NN + rowoff + i;
    KlT[i][c] = Kg[g]; Vdl[c][i] = Vdg[g]; Vul[c][i] = Vug[g];
  }
  __syncthreads();

  int lane = tid & 31, wv = tid >> 5;
  int lr = lane & 15, lhA = (lane >> 4) << 3, lhB = (lane >> 4) << 3, mo = lhA;
  float sc = scale_w[0];

  {  // scores
    int i = wv * 16 + lr;
    v16h a;
#pragma unroll
    for (int t = 0; t < 8; ++t) a[t]     = Qg[(size_t)(lhA + t) * NN + rowoff + i];
#pragma unroll
    for (int t = 0; t < 8; ++t) a[8 + t] = Qg[(size_t)(16 + lhA + t) * NN + rowoff + i];
#pragma unroll
    for (int jt = 0; jt < 8; ++jt) {
      int j = jt * 16 + lr;
      v16h bf = fragB(&KlT[j][lhB]);
      v8f acc = {};
      acc = wmma_f16(a, bf, acc);
      v8h sv;
#pragma unroll
      for (int v = 0; v < 8; ++v) sv[v] = (_Float16)(acc[v] * sc);
      *(v8h*)&SlT[j][wv * 16 + mo] = sv;
#pragma unroll
      for (int v = 0; v < 8; ++v) Sl[wv * 16 + v + mo][j] = sv[v];
    }
  }
  __syncthreads();

  if (tid < 128) {  // row stats (softmax over j) from Sl rows
    float m = -1e30f;
    for (int j = 0; j < 128; j += 8) {
      v8h s = *(const v8h*)&Sl[tid][j];
#pragma unroll
      for (int t = 0; t < 8; ++t) m = fmaxf(m, (float)s[t]);
    }
    float s = 0.f;
    for (int j = 0; j < 128; j += 8) {
      v8h sv = *(const v8h*)&Sl[tid][j];
#pragma unroll
      for (int t = 0; t < 8; ++t) s += __expf((float)sv[t] - m);
    }
    rmax[tid] = m; rinv[tid] = 1.f / s;
  } else {          // col stats (softmax over i) from SlT rows
    int j = tid - 128;
    float m = -1e30f;
    for (int i = 0; i < 128; i += 8) {
      v8h s = *(const v8h*)&SlT[j][i];
#pragma unroll
      for (int t = 0; t < 8; ++t) m = fmaxf(m, (float)s[t]);
    }
    float s = 0.f;
    for (int i = 0; i < 128; i += 8) {
      v8h sv = *(const v8h*)&SlT[j][i];
#pragma unroll
      for (int t = 0; t < 8; ++t) s += __expf((float)sv[t] - m);
    }
    cmax[j] = m; cinv[j] = 1.f / s;
  }
  __syncthreads();

  {  // d2u_w[c,i] = sum_j softmax_row(S)[i,j] Vd[c,j]
    v8f acc[2] = {};
    int i = wv * 16 + lr;
    float m = rmax[i], inv = rinv[i];
#pragma unroll
    for (int kk = 0; kk < 128; kk += 32) {
      v16h a = fragP(&Sl[i][kk + lhA], m, inv);
#pragma unroll
      for (int ct = 0; ct < 2; ++ct) {
        v16h bf = fragB(&Vdl[ct * 16 + lr][kk + lhB]);
        acc[ct] = wmma_f16(a, bf, acc[ct]);
      }
    }
    float* og = d2u + ((size_t)b * NC + 32) * NN;
#pragma unroll
    for (int ct = 0; ct < 2; ++ct) {
      int c = ct * 16 + lr;
#pragma unroll
      for (int v = 0; v < 8; ++v) {
        int ii = wv * 16 + v + mo;
        og[(size_t)c * NN + rowoff + ii] = acc[ct][v];
      }
    }
  }

  {  // u2d_w[c,j] = sum_i softmax_col(S)[i,j] Vu[c,i]  (P2 fragments from SlT)
    v8f acc[2] = {};
    int j = wv * 16 + lr;
    float m = cmax[j], inv = cinv[j];
#pragma unroll
    for (int kk = 0; kk < 128; kk += 32) {
      v16h a = fragP(&SlT[j][kk + lhA], m, inv);
#pragma unroll
      for (int ct = 0; ct < 2; ++ct) {
        v16h bf = fragB(&Vul[ct * 16 + lr][kk + lhB]);
        acc[ct] = wmma_f16(a, bf, acc[ct]);
      }
    }
    float* og = u2d + ((size_t)b * NC + 32) * NN;
#pragma unroll
    for (int ct = 0; ct < 2; ++ct) {
      int c = ct * 16 + lr;
#pragma unroll
      for (int v = 0; v < 8; ++v) {
        int jj = wv * 16 + v + mo;
        og[(size_t)c * NN + rowoff + jj] = acc[ct][v];
      }
    }
  }
}

// ===========================================================================
// Kernel 6: channel attention scores: A_c[b] = softmax(sc * q_c k_c^T) [32x32]
// Gram over n = 16384; per-wave WMMA partials reduced via ds_add_f32.
// Fragments load straight from global as b128 (rows are contiguous in n).
// ===========================================================================
__global__ __launch_bounds__(256) void k_attn_c_scores(
    const _Float16* __restrict__ qkv_u, const _Float16* __restrict__ qkv_d,
    const float* __restrict__ scale_c, _Float16* __restrict__ Ac) {
  int b = blockIdx.x;
  const _Float16* qg = qkv_u + ((size_t)b * NOC + 64) * NN;   // q_c
  const _Float16* kg = qkv_d + ((size_t)b * NOC + 64) * NN;   // k_c
  __shared__ float Sacc[32][33];
  int tid = threadIdx.x;
  for (int idx = tid; idx < 32 * 33; idx += 256) (&Sacc[0][0])[idx] = 0.f;
  __syncthreads();

  int lane = tid & 31, wv = tid >> 5;
  int lr = lane & 15, lhA = (lane >> 4) << 3, lhB = (lane >> 4) << 3, mo = lhA;

  v8f acc[2][2] = {};
  int n0 = wv * 2048;
  for (int kb = n0; kb < n0 + 2048; kb += 32) {
    v16h a[2], bf[2];
#pragma unroll
    for (int ct = 0; ct < 2; ++ct) a[ct]  = fragA(qg + (size_t)(ct * 16 + lr) * NN + kb + lhA);
#pragma unroll
    for (int dt = 0; dt < 2; ++dt) bf[dt] = fragB(kg + (size_t)(dt * 16 + lr) * NN + kb + lhB);
#pragma unroll
    for (int ct = 0; ct < 2; ++ct)
#pragma unroll
      for (int dt = 0; dt < 2; ++dt)
        acc[ct][dt] = wmma_f16(a[ct], bf[dt], acc[ct][dt]);
  }
#pragma unroll
  for (int ct = 0; ct < 2; ++ct)
#pragma unroll
    for (int dt = 0; dt < 2; ++dt)
#pragma unroll
      for (int v = 0; v < 8; ++v)
        atomicAdd(&Sacc[ct * 16 + v + mo][dt * 16 + lr], acc[ct][dt][v]);
  __syncthreads();

  if (tid < 32) {
    float sc = scale_c[0];
    float m = -1e30f;
    for (int d = 0; d < 32; ++d) m = fmaxf(m, Sacc[tid][d] * sc);
    float s = 0.f;
    float e[32];
    for (int d = 0; d < 32; ++d) { e[d] = __expf(Sacc[tid][d] * sc - m); s += e[d]; }
    float inv = 1.f / s;
    for (int d = 0; d < 32; ++d)
      Ac[((size_t)b * 32 + tid) * 32 + d] = (_Float16)(e[d] * inv);
  }
}

// ===========================================================================
// Kernel 7: apply channel attention: out[c,n] = A_c @ V  (both vd and vu).
// V tiles stored TRANSPOSED in LDS so B-fragments are contiguous.
// ===========================================================================
__global__ __launch_bounds__(256) void k_attn_c_apply(
    const _Float16* __restrict__ qkv_u, const _Float16* __restrict__ qkv_d,
    const _Float16* __restrict__ Ac,
    float* __restrict__ d2u, float* __restrict__ u2d) {
  int b = blockIdx.y;
  int n0 = blockIdx.x * 256;
  const _Float16* vdg = qkv_d + ((size_t)b * NOC + 160) * NN;  // vd_c
  const _Float16* vug = qkv_u + ((size_t)b * NOC + 160) * NN;  // vu_c

  __shared__ __align__(16) _Float16 Al [32][32];    // [c][d]
  __shared__ __align__(16) _Float16 VdT[256][32];   // [n][d]
  __shared__ __align__(16) _Float16 VuT[256][32];   // [n][d]

  int tid = threadIdx.x;
  for (int idx = tid; idx < 32 * 32; idx += 256)
    Al[idx >> 5][idx & 31] = Ac[(size_t)b * 1024 + idx];
  for (int idx = tid; idx < 32 * 256; idx += 256) {
    int r = idx >> 8, n = idx & 255;
    VdT[n][r] = vdg[(size_t)r * NN + n0 + n];
    VuT[n][r] = vug[(size_t)r * NN + n0 + n];
  }
  __syncthreads();

  int lane = tid & 31, wv = tid >> 5;
  int lr = lane & 15, lhA = (lane >> 4) << 3, lhB = (lane >> 4) << 3, mo = lhA;

  v16h aF[2];
  aF[0] = fragA(&Al[lr][lhA]);
  aF[1] = fragA(&Al[16 + lr][lhA]);

  float* o1 = d2u + ((size_t)b * NC + 64) * NN;
  float* o2 = u2d + ((size_t)b * NC + 64) * NN;
#pragma unroll
  for (int nt2 = 0; nt2 < 2; ++nt2) {
    int nt = wv * 2 + nt2;
    v16h bD = fragB(&VdT[nt * 16 + lr][lhB]);
    v16h bU = fragB(&VuT[nt * 16 + lr][lhB]);
#pragma unroll
    for (int ct = 0; ct < 2; ++ct) {
      v8f accD = {}, accU = {};
      accD = wmma_f16(aF[ct], bD, accD);
      accU = wmma_f16(aF[ct], bU, accU);
      int n = n0 + nt * 16 + lr;
#pragma unroll
      for (int v = 0; v < 8; ++v) {
        int c = ct * 16 + v + mo;
        o1[(size_t)c * NN + n] = accD[v];
        o2[(size_t)c * NN + n] = accU[v];
      }
    }
  }
}

// ===========================================================================
// Kernel 8: epilogue — rescale/rebias + residuals, float4 streaming
// ===========================================================================
__global__ __launch_bounds__(256) void k_epilogue(
    const float* __restrict__ x_u, const float* __restrict__ x_d,
    const float* __restrict__ d2u, const float* __restrict__ u2d,
    const float* __restrict__ resc, float* __restrict__ out) {
  size_t total4 = (size_t)NB * NC * NN / 4;
  const float4* xu4 = (const float4*)x_u;
  const float4* xd4 = (const float4*)x_d;
  const float4* a4  = (const float4*)d2u;
  const float4* b4  = (const float4*)u2d;
  float4* o4 = (float4*)out;
  for (size_t i4 = (size_t)blockIdx.x * blockDim.x + threadIdx.x; i4 < total4;
       i4 += (size_t)gridDim.x * blockDim.x) {
    size_t bc = (i4 * 4) / NN;
    int b = (int)(bc / NC), c = (int)(bc % NC);
    float ru = resc[((size_t)(0 * 2 + 0) * NB + b) * NC + c];
    float bu = resc[((size_t)(0 * 2 + 1) * NB + b) * NC + c];
    float rd = resc[((size_t)(1 * 2 + 0) * NB + b) * NC + c];
    float bd = resc[((size_t)(1 * 2 + 1) * NB + b) * NC + c];
    float4 xa = xu4[i4], xb = xd4[i4], da = a4[i4], db = b4[i4];
    float4 o;
    o.x = da.x * ru + bu + xa.x + db.x * rd + bd + xb.x;
    o.y = da.y * ru + bu + xa.y + db.y * rd + bd + xb.y;
    o.z = da.z * ru + bu + xa.z + db.z * rd + bd + xb.z;
    o.w = da.w * ru + bu + xa.w + db.w * rd + bd + xb.w;
    o4[i4] = o;
  }
}

// ===========================================================================
extern "C" void kernel_launch(void* const* d_in, const int* in_sizes, int n_in,
                              void* d_out, int out_size, void* d_ws, size_t ws_size,
                              hipStream_t stream) {
  (void)in_sizes; (void)n_in; (void)out_size; (void)ws_size;
  const float* x_u   = (const float*)d_in[0];
  const float* x_d   = (const float*)d_in[1];
  const float* nw_u  = (const float*)d_in[2];
  const float* nb_u  = (const float*)d_in[3];
  const float* m1w_u = (const float*)d_in[4];
  const float* m1b_u = (const float*)d_in[5];
  const float* m2w_u = (const float*)d_in[6];
  const float* m2b_u = (const float*)d_in[7];
  const float* nw_d  = (const float*)d_in[8];
  const float* nb_d  = (const float*)d_in[9];
  const float* m1w_d = (const float*)d_in[10];
  const float* m1b_d = (const float*)d_in[11];
  const float* m2w_d = (const float*)d_in[12];
  const float* m2b_d = (const float*)d_in[13];
  const float* Wq_u  = (const float*)d_in[14];
  const float* bq_u  = (const float*)d_in[15];
  const float* Wq_d  = (const float*)d_in[16];
  const float* bq_d  = (const float*)d_in[17];
  const float* sc_h  = (const float*)d_in[18];
  const float* sc_w  = (const float*)d_in[19];
  const float* sc_c  = (const float*)d_in[20];

  char* ws = (char*)d_ws;
  float*     stats  = (float*)(ws + OFF_STATS);
  _Float16*  Wh     = (_Float16*)(ws + OFF_WH);
  float*     biasQ  = (float*)(ws + OFF_BIASQ);
  float*     resc   = (float*)(ws + OFF_RESC);
  _Float16*  Ac     = (_Float16*)(ws + OFF_AC);
  _Float16*  qkv_u  = (_Float16*)(ws + OFF_QKVU);
  _Float16*  qkv_d  = (_Float16*)(ws + OFF_QKVD);
  float*     d2u    = (float*)(ws + OFF_D2U);
  float*     u2d    = (float*)(ws + OFF_U2D);

  k_stats<<<16, 256, 0, stream>>>(x_u, x_d, stats);
  k_prep<<<2, 192, 0, stream>>>(Wq_u, bq_u, Wq_d, bq_d,
                                nw_u, nb_u, m1w_u, m1b_u, m2w_u, m2b_u,
                                nw_d, nb_d, m1w_d, m1b_d, m2w_d, m2b_d,
                                stats, Wh, biasQ, resc);
  dim3 gC(NN / 128, NOC / 64, 16);
  k_qkv_gemm<<<gC, 256, 0, stream>>>(x_u, x_d, Wh, biasQ, stats, qkv_u, qkv_d);
  k_attn_h<<<NB * NW, 256, 0, stream>>>(qkv_u, qkv_d, sc_h, d2u, u2d);
  k_attn_w<<<NB * NH, 256, 0, stream>>>(qkv_u, qkv_d, sc_w, d2u, u2d);
  k_attn_c_scores<<<NB, 256, 0, stream>>>(qkv_u, qkv_d, sc_c, Ac);
  dim3 gF2(NN / 256, NB);
  k_attn_c_apply<<<gF2, 256, 0, stream>>>(qkv_u, qkv_d, Ac, d2u, u2d);
  k_epilogue<<<2048, 256, 0, stream>>>(x_u, x_d, d2u, u2d, resc, (float*)d_out);
}
